// FlightDelayGNN_27719718928541
// MI455X (gfx1250) — matrix operand order repaired and checked by reading, not compiled, not run on hardware
//
#include <hip/hip_runtime.h>
#include <cstddef>
#include <cstdint>

// ---------------------------------------------------------------------------
// CDNA5 (gfx1250) implementation of the FlightDelayGNN forward pass.
// Dense linears: v_wmma_f32_16x16x32_bf16, 4-wave blocks, B tile staged into
// LDS with global_load_async_to_lds_b128 (ASYNCcnt) and read via ds_load_b128.
// Edge softmax/aggregation: f32 global atomics (resolve in the 192MB L2).
// Elementwise/LN/GRU: wave32 VALU kernels.
// ---------------------------------------------------------------------------

typedef __bf16 bf16;
typedef __attribute__((ext_vector_type(16))) __bf16   v16bf;
typedef __attribute__((ext_vector_type(8)))  float    v8f;
typedef __attribute__((ext_vector_type(4)))  unsigned u32x4;

#define ACT_NONE     0
#define ACT_RELU     1
#define ACT_SOFTPLUS 5

__device__ __forceinline__ float sigf(float x) { return 1.0f / (1.0f + __expf(-x)); }

__device__ __forceinline__ float actf(float x, int a) {
  if (a == ACT_RELU)     return fmaxf(x, 0.0f);
  if (a == ACT_SOFTPLUS) return (x > 20.0f) ? x : log1pf(__expf(x));
  return x;
}

// wave32 reduction (gfx1250 is wave32-only)
__device__ __forceinline__ float wsum(float v) {
#pragma unroll
  for (int m = 16; m >= 1; m >>= 1) v += __shfl_xor(v, m, 32);
  return v;
}

__device__ __forceinline__ void atomicMaxF(float* addr, float val) {
  unsigned* ua = (unsigned*)addr;
  unsigned cur = *ua;
  while (__uint_as_float(cur) < val) {
    unsigned old = atomicCAS(ua, cur, __float_as_uint(val));
    if (old == cur) break;
    cur = old;
  }
}

// ---------------------------------------------------------------------------
// WMMA GEMM:  C[M,N] = act( A[M,K](bf16,row-major) * W[N,K]^T (bf16) + bias )
// Block = 128 threads (4 waves) -> 64(M) x 64(N) tile. Per 32-wide K step:
//  * all 128 lanes stage the shared 64x32 bf16 W tile into LDS with
//    global_load_async_to_lds_b128 (ASYNCcnt), s_wait_asynccnt, barrier;
//  * each active wave loads its private A fragment (2 x b128, ISA 7.12.2:
//    lane = M row, VGPR v in 0..3 -> K = half*8 + 2v ; v in 4..7 -> +16),
//    then all 4 B fragments from LDS (8 x ds_load_b128 issued back-to-back),
//    then 4 wmma (dscnt drains incrementally).
// CM / CN: bounds handling only when M%16 / N%64 are ragged. nsub is a
// compile-time 4 on the fast path; waveActive is hoisted by loop splitting
// (staging + barrier loop for tail waves keeps the barrier count identical).
// ---------------------------------------------------------------------------
template <bool CM, bool CN>
__global__ __launch_bounds__(128)
void gemm_wmma_bf16(const bf16* __restrict__ A, int lda,
                    const bf16* __restrict__ W, int ldw,
                    const float* __restrict__ bias,
                    float* __restrict__ C, int ldc,
                    bf16* __restrict__ Cb, int ldcb,
                    int M, int N, int K, int act) {
  __shared__ __align__(16) bf16 Bs[64 * 32];

  const int tid  = threadIdx.x;
  const int lane = tid & 31;
  const int l15  = lane & 15;
  const int half = lane >> 4;
  const int wv   = tid >> 5;
  const int m0   = blockIdx.x * 64 + wv * 16;
  const int n0   = blockIdx.y * 64;
  const bool waveActive = (m0 < M);       // uniform per wave (M % 16 == 0 or CM)
  const int mrow = m0 + l15;
  const bool mok = CM ? (mrow < M) : true;

  int nsub = 4;                            // constant 4 on the !CN fast path
  if (CN) { int t = (N - n0 + 15) >> 4; nsub = (t < 4) ? t : 4; }

  // staging: thread t loads 2x16B -> LDS; row = t>>1 (0..63), k half = t&1
  const int srow   = tid >> 1;
  const int schunk = tid & 1;
  const unsigned ldsDst = (unsigned)(uintptr_t)(&Bs[0]) + (unsigned)tid * 32u;
  const bool sok = CN ? (n0 + srow < N) : true;
  const bf16* wsrc = W + (size_t)(n0 + srow) * ldw + schunk * 16;
  const bf16* arow = A + (size_t)mrow * lda;

  union FragU { u32x4 q[2]; v16bf v; };
  const u32x4 zq = {0u, 0u, 0u, 0u};

  auto stage = [&](int k) {
    if (sok) {
      uint64_t ga = (uint64_t)(uintptr_t)(wsrc + k);
      asm volatile("global_load_async_to_lds_b128 %0, %1, off"
                   :: "v"(ldsDst), "v"(ga) : "memory");
      asm volatile("global_load_async_to_lds_b128 %0, %1, off"
                   :: "v"(ldsDst + 16u), "v"(ga + 16) : "memory");
    }
    asm volatile("s_wait_asynccnt 0x0" ::: "memory");
  };

  v8f acc[4];
#pragma unroll
  for (int s = 0; s < 4; s++)
#pragma unroll
    for (int r = 0; r < 8; r++) acc[s][r] = 0.0f;

  if (waveActive) {
    for (int k = 0; k < K; k += 32) {
      stage(k);
      __syncthreads();
      FragU a;
      if (!CM || mok) {
        const bf16* ap = arow + k;
        a.q[0] = *(const u32x4*)(ap + half * 8);
        a.q[1] = *(const u32x4*)(ap + 16 + half * 8);
      } else {
        a.q[0] = zq; a.q[1] = zq;
      }
      __builtin_prefetch(arow + k + 32, 0, 3);   // speculative; no EXEC change
      FragU bfr[4];
      for (int s = 0; s < nsub; s++) {
        const bf16* bsrc = &Bs[(s * 16 + l15) * 32 + half * 8];
        bfr[s].q[0] = *(const u32x4*)(bsrc);       // ds_load_b128
        bfr[s].q[1] = *(const u32x4*)(bsrc + 16);
      }
      for (int s = 0; s < nsub; s++)
        acc[s] = __builtin_amdgcn_wmma_f32_16x16x32_bf16(
            false, a.v, false, bfr[s].v, (short)0, acc[s], false, false);
      __syncthreads();
    }
  } else {
    // tail waves: keep staging duty + identical barrier count
    for (int k = 0; k < K; k += 32) {
      stage(k);
      __syncthreads();
      __syncthreads();
    }
    return;
  }

  for (int s = 0; s < nsub; s++) {
    int ncol = n0 + s * 16 + l15;
    bool nok = CN ? (ncol < N) : true;
    float bv = (bias != nullptr && nok) ? bias[ncol] : 0.0f;
#pragma unroll
    for (int r = 0; r < 8; r++) {
      int row = m0 + half * 8 + r;        // D layout: M = r + 8*(lane>>4)
      bool rok = CM ? (row < M) : true;
      if (rok && nok) {
        float v = actf(acc[s][r] + bv, act);
        if (C)  C[(size_t)row * ldc + ncol] = v;
        if (Cb) Cb[(size_t)row * ldcb + ncol] = (bf16)v;
      }
    }
  }
}

// ------------------------------ elementwise --------------------------------

__global__ void k_cvt_bf16(const float* __restrict__ in, bf16* __restrict__ out, long n) {
  long i = (long)blockIdx.x * blockDim.x + threadIdx.x;
  if (i < n) out[i] = (bf16)in[i];
}

__global__ void k_fill(float* __restrict__ p, float v, long n) {
  long i = (long)blockIdx.x * blockDim.x + threadIdx.x;
  if (i < n) p[i] = v;
}

// in: [7][4][32][32] f32 (d,e) -> out: bf16 (e,d) per (rel,head). grid=28, block=(32,32)
__global__ void k_rel_transpose(const float* __restrict__ in, bf16* __restrict__ out) {
  int blk = blockIdx.x;
  const float* ip = in + (size_t)blk * 1024;
  bf16* op = out + (size_t)blk * 1024;
  op[threadIdx.y * 32 + threadIdx.x] = (bf16)ip[threadIdx.x * 32 + threadIdx.y];
}

// one wave per row of 128
__global__ __launch_bounds__(32)
void k_ln(const float* __restrict__ x, const float* __restrict__ g, const float* __restrict__ b,
          float* __restrict__ out, bf16* __restrict__ outb) {
  int row = blockIdx.x, lane = threadIdx.x;
  const float* xr = x + (size_t)row * 128;
  float v[4]; float s = 0.0f;
#pragma unroll
  for (int i = 0; i < 4; i++) { v[i] = xr[lane + 32 * i]; s += v[i]; }
  s = wsum(s) * (1.0f / 128.0f);
  float var = 0.0f;
#pragma unroll
  for (int i = 0; i < 4; i++) { float d = v[i] - s; var += d * d; }
  var = wsum(var) * (1.0f / 128.0f);
  float inv = rsqrtf(var + 1e-5f);
#pragma unroll
  for (int i = 0; i < 4; i++) {
    int c = lane + 32 * i;
    float y = (v[i] - s) * inv * g[c] + b[c];
    if (out)  out[(size_t)row * 128 + c] = y;
    if (outb) outb[(size_t)row * 128 + c] = (bf16)y;
  }
}

__global__ void k_rot_gate(const float* __restrict__ ea, const float* __restrict__ w1,
                           const float* __restrict__ b1, const float* __restrict__ w2,
                           const float* __restrict__ b2, float* __restrict__ g, int n) {
  int e = blockIdx.x * blockDim.x + threadIdx.x;
  if (e >= n) return;
  float x0 = ea[2 * e], x1 = ea[2 * e + 1];
  float s = b2[0];
#pragma unroll 4
  for (int j = 0; j < 32; j++) {
    float h = fmaxf(w1[2 * j] * x0 + w1[2 * j + 1] * x1 + b1[j], 0.0f);
    s += w2[j] * h;
  }
  g[e] = sigf(s);
}

__global__ void k_rot_scatter(float* __restrict__ dst, const float* __restrict__ src,
                              const float* __restrict__ g, const int* __restrict__ es,
                              const int* __restrict__ ed, int E) {
  long t = (long)blockIdx.x * blockDim.x + threadIdx.x;
  if (t >= (long)E * 128) return;
  int e = (int)(t >> 7), i = (int)(t & 127);
  atomicAdd(&dst[(size_t)ed[e] * 128 + i], g[e] * src[(size_t)es[e] * 128 + i]);
}

__global__ void k_edge_logit(const int* __restrict__ es, const int* __restrict__ ed,
                             const float* __restrict__ Q, const float* __restrict__ KT,
                             const float* __restrict__ prel, float scale,
                             float* __restrict__ lg, float* __restrict__ mx, int E) {
  int t = blockIdx.x * blockDim.x + threadIdx.x;
  if (t >= E * 4) return;
  int e = t >> 2, h = t & 3;
  int s = es[e], d = ed[e];
  const float* q = Q + (size_t)d * 128 + h * 32;
  const float* k = KT + (size_t)s * 128 + h * 32;
  float a = 0.0f;
#pragma unroll 8
  for (int i = 0; i < 32; i++) a += q[i] * k[i];
  a *= prel[h] * scale;
  lg[t] = a;
  atomicMaxF(&mx[d * 4 + h], a);
}

__global__ void k_mx_fix(float* __restrict__ mx, int n) {
  int i = blockIdx.x * blockDim.x + threadIdx.x;
  if (i < n && mx[i] < -1e37f) mx[i] = 0.0f;
}

__global__ void k_edge_exp(const int* __restrict__ ed, float* __restrict__ lg,
                           const float* __restrict__ mx, float* __restrict__ den, int E) {
  int t = blockIdx.x * blockDim.x + threadIdx.x;
  if (t >= E * 4) return;
  int e = t >> 2, h = t & 3;
  int d = ed[e];
  float v = __expf(lg[t] - mx[d * 4 + h]);
  lg[t] = v;
  atomicAdd(&den[d * 4 + h], v);
}

__global__ void k_edge_agg(const int* __restrict__ es, const int* __restrict__ ed,
                           const float* __restrict__ lg, const float* __restrict__ den,
                           const float* __restrict__ VT, float* __restrict__ agg, int E) {
  long t = (long)blockIdx.x * blockDim.x + threadIdx.x;
  if (t >= (long)E * 128) return;
  int e = (int)(t >> 7), i = (int)(t & 127), h = i >> 5;
  int s = es[e], d = ed[e];
  float a = lg[e * 4 + h] / fmaxf(den[d * 4 + h], 1e-16f);
  atomicAdd(&agg[(size_t)d * 128 + i], a * VT[(size_t)s * 128 + i]);
}

__global__ void k_gelu_cvt(const float* __restrict__ in, bf16* __restrict__ out, long n) {
  long i = (long)blockIdx.x * blockDim.x + threadIdx.x;
  if (i >= n) return;
  float x = in[i];
  out[i] = (bf16)(0.5f * x * (1.0f + erff(x * 0.70710678118f)));
}

// x <- LN( (a*o + (1-a)*x) + x ), a = sigmoid(skip); writes f32+bf16 in place
__global__ __launch_bounds__(32)
void k_skip_res_ln(const float* __restrict__ o, float* __restrict__ x, bf16* __restrict__ xb,
                   const float* __restrict__ skip, const float* __restrict__ g,
                   const float* __restrict__ b) {
  int row = blockIdx.x, lane = threadIdx.x;
  float a = sigf(skip[0]);
  float v[4]; float s = 0.0f;
#pragma unroll
  for (int i = 0; i < 4; i++) {
    int c = lane + 32 * i;
    float xo = x[(size_t)row * 128 + c];
    float t = a * o[(size_t)row * 128 + c] + (1.0f - a) * xo + xo;
    v[i] = t; s += t;
  }
  s = wsum(s) * (1.0f / 128.0f);
  float var = 0.0f;
#pragma unroll
  for (int i = 0; i < 4; i++) { float d = v[i] - s; var += d * d; }
  var = wsum(var) * (1.0f / 128.0f);
  float inv = rsqrtf(var + 1e-5f);
#pragma unroll
  for (int i = 0; i < 4; i++) {
    int c = lane + 32 * i;
    float y = (v[i] - s) * inv * g[c] + b[c];
    x[(size_t)row * 128 + c] = y;
    xb[(size_t)row * 128 + c] = (bf16)y;
  }
}

__global__ void k_gru(const float* __restrict__ gi, const float* __restrict__ gh,
                      const float* __restrict__ hold, float* __restrict__ hnew,
                      bf16* __restrict__ hb, int rows, int H) {
  long t = (long)blockIdx.x * blockDim.x + threadIdx.x;
  if (t >= (long)rows * H) return;
  int row = (int)(t / H), j = (int)(t % H);
  const float* a = gi + (size_t)row * 3 * H;
  const float* c = gh + (size_t)row * 3 * H;
  float r = sigf(a[j] + c[j]);
  float z = sigf(a[H + j] + c[H + j]);
  float n = tanhf(a[2 * H + j] + r * c[2 * H + j]);
  float h = (1.0f - z) * n + z * hold[t];
  hnew[t] = h;
  if (hb) hb[t] = (bf16)h;
}

__global__ void k_gather_rows(float* __restrict__ dst, const float* __restrict__ src,
                              const int* __restrict__ idx, int cols, int n) {
  long t = (long)blockIdx.x * blockDim.x + threadIdx.x;
  if (t >= (long)n * cols) return;
  int r = (int)(t / cols), c = (int)(t % cols);
  dst[t] = src[(size_t)idx[r] * cols + c];
}

__global__ void k_scatter_rows(float* __restrict__ dst, const int* __restrict__ didx,
                               const float* __restrict__ src, const int* __restrict__ sidx,
                               int cols, int n) {
  long t = (long)blockIdx.x * blockDim.x + threadIdx.x;
  if (t >= (long)n * cols) return;
  int r = (int)(t / cols), c = (int)(t % cols);
  int dr = didx ? didx[r] : r;
  int sr = sidx ? sidx[r] : r;
  dst[(size_t)dr * cols + c] = src[(size_t)sr * cols + c];
}

__global__ void k_concat_cvt(const float* __restrict__ emb, const float* __restrict__ apc,
                             const float* __restrict__ tc, bf16* __restrict__ out, int n) {
  long t = (long)blockIdx.x * blockDim.x + threadIdx.x;
  if (t >= (long)n * 384) return;
  int r = (int)(t / 384), c = (int)(t % 384);
  float v = (c < 128) ? emb[(size_t)r * 128 + c]
          : (c < 256) ? apc[(size_t)r * 128 + (c - 128)]
                      : tc[(size_t)r * 128 + (c - 256)];
  out[t] = (bf16)v;
}

// out = sigmoid(gate) * relu(LN(fuse)) + emb
__global__ __launch_bounds__(32)
void k_flout(const float* __restrict__ fu, const float* __restrict__ ga,
             const float* __restrict__ emb, const float* __restrict__ g,
             const float* __restrict__ b, float* __restrict__ out, bf16* __restrict__ outb) {
  int row = blockIdx.x, lane = threadIdx.x;
  float v[4]; float s = 0.0f;
#pragma unroll
  for (int i = 0; i < 4; i++) { v[i] = fu[(size_t)row * 128 + lane + 32 * i]; s += v[i]; }
  s = wsum(s) * (1.0f / 128.0f);
  float var = 0.0f;
#pragma unroll
  for (int i = 0; i < 4; i++) { float d = v[i] - s; var += d * d; }
  var = wsum(var) * (1.0f / 128.0f);
  float inv = rsqrtf(var + 1e-5f);
#pragma unroll
  for (int i = 0; i < 4; i++) {
    int c = lane + 32 * i;
    float ln = (v[i] - s) * inv * g[c] + b[c];
    float y = sigf(ga[(size_t)row * 128 + c]) * fmaxf(ln, 0.0f) + emb[(size_t)row * 128 + c];
    out[(size_t)row * 128 + c] = y;
    outb[(size_t)row * 128 + c] = (bf16)y;
  }
}

__global__ void k_final(const float* __restrict__ p0, const float* __restrict__ lg,
                        const float* __restrict__ up, float* __restrict__ out, int n) {
  int i = blockIdx.x * blockDim.x + threadIdx.x;
  if (i >= n) return;
  out[i] = p0[i] + sigf(lg[(size_t)i * 6 + 4]) * up[2 * i]
                 + sigf(lg[(size_t)i * 6 + 5]) * up[2 * i + 1];
}

// ---------------------------------------------------------------------------

extern "C" void kernel_launch(void* const* d_in, const int* in_sizes, int n_in,
                              void* d_out, int out_size, void* d_ws, size_t ws_size,
                              hipStream_t stream) {
  (void)in_sizes; (void)n_in; (void)out_size; (void)ws_size;

  const long NAP = 3000, NFL = 100000, NTL = 5000, EROT = 90000;

  const float* ap_x       = (const float*)d_in[0];
  const float* fl_x       = (const float*)d_in[1];
  const float* rot_ea     = (const float*)d_in[2];
  const float* ap_h_in    = (const float*)d_in[3];
  const float* tail_table = (const float*)d_in[4];
  const float* Pf         = (const float*)d_in[5];
  const int*   ei_rot_ff  = (const int*)d_in[11];
  const int*   ei_dep_fr  = (const int*)d_in[12];
  const int*   tail_id    = (const int*)d_in[14];

  // ---- param offsets (jax tree_flatten: dict keys sorted, lists in order) ----
  long o = 0;
  auto T = [&](long n) { long r = o; o += n; return r; };
  const long ap_ctx_b = T(128),  ap_ctx_w = T(16384);
  long ap_ln_b[2], ap_ln_g[2];
  for (int l = 0; l < 2; l++) { ap_ln_b[l] = T(128); ap_ln_g[l] = T(128); }
  const long ap_proj_b = T(128), ap_proj_w = T(128 * 32);
  const long ap_proj_ln_b = T(128), ap_proj_ln_g = T(128);
  const long cls1_b = T(128), cls1_w = T(16384);
  const long cls2_b = T(6),   cls2_w = T(768);
  struct ConvOff {
    long a_rel[7], k_b[2], k_w[2], m_rel[7], out_b[2], out_w[2],
         p_rel[7], q_b[2], q_w[2], skip[2], v_b[2], v_w[2];
  } co[2];
  for (int L = 0; L < 2; L++) {
    for (int r = 0; r < 7; r++) co[L].a_rel[r] = T(4096);
    for (int t = 0; t < 2; t++) { co[L].k_b[t] = T(128); co[L].k_w[t] = T(16384); }
    for (int r = 0; r < 7; r++) co[L].m_rel[r] = T(4096);
    for (int t = 0; t < 2; t++) { co[L].out_b[t] = T(128); co[L].out_w[t] = T(16384); }
    for (int r = 0; r < 7; r++) co[L].p_rel[r] = T(4);
    for (int t = 0; t < 2; t++) { co[L].q_b[t] = T(128); co[L].q_w[t] = T(16384); }
    for (int t = 0; t < 2; t++) co[L].skip[t] = T(1);
    for (int t = 0; t < 2; t++) { co[L].v_b[t] = T(128); co[L].v_w[t] = T(16384); }
  }
  long fl_ln_b[2], fl_ln_g[2];
  for (int l = 0; l < 2; l++) { fl_ln_b[l] = T(128); fl_ln_g[l] = T(128); }
  const long fl_proj_b = T(128), fl_proj_w = T(8192);
  const long fl_proj_ln_b = T(128), fl_proj_ln_g = T(128);
  const long fuse_b = T(128), fuse_w = T(49152);
  const long fuse_ln_b = T(128), fuse_ln_g = T(128);
  const long gate_b = T(128), gate_w = T(49152);
  long gru_bhh[2], gru_bih[2], gru_whh[2], gru_wih[2];
  for (int l = 0; l < 2; l++) {
    gru_bhh[l] = T(384); gru_bih[l] = T(384); gru_whh[l] = T(49152); gru_wih[l] = T(49152);
  }
  const long head1_b = T(256), head1_w = T(32768);
  const long head2_b = T(128), head2_w = T(32768);
  const long head3_b = T(1),   head3_w = T(128);
  const long rot1_b = T(32), rot1_w = T(64);
  const long rot2_b = T(1),  rot2_w = T(32);
  const long tg_bhh = T(192), tg_bih = T(192), tg_whh = T(12288), tg_wih = T(24576);
  const long tp_b = T(128), tp_w = T(8192);
  const long up1_b = T(128), up1_w = T(16384);
  const long up2_b = T(2),   up2_w = T(256);

  // ---- workspace bump allocator (256B aligned slabs) ----
  size_t wo = 0;
  char* WSB = (char*)d_ws;
  auto AL  = [&](size_t bytes) -> void* {
    bytes = (bytes + 255) & ~(size_t)255;
    void* p = WSB + wo; wo += bytes; return p;
  };
  auto ALF = [&](long n) -> float* { return (float*)AL((size_t)n * 4); };
  auto ALB = [&](long n) -> bf16*  { return (bf16*)AL((size_t)n * 2); };
  auto nb  = [](long n) -> unsigned { return (unsigned)((n + 255) / 256); };
  auto WB  = [&](long off, long n) -> bf16* {   // bf16 copy of a weight, aligned
    bf16* p = ALB(n);
    k_cvt_bf16<<<nb(n), 256, 0, stream>>>(Pf + off, p, n);
    return p;
  };
  auto GEMM = [&](const bf16* Ax, int lda, const bf16* Wx, int ldw, const float* bias,
                  float* Cx, int ldc, bf16* Cbx, int ldcb, int M, int N, int K, int act) {
    dim3 grid((unsigned)((M + 63) / 64), (unsigned)((N + 63) / 64));
    const bool cm = (M & 15) != 0;
    const bool cn = (N & 63) != 0;
    if (!cm && !cn)
      gemm_wmma_bf16<false, false><<<grid, 128, 0, stream>>>(Ax, lda, Wx, ldw, bias, Cx, ldc, Cbx, ldcb, M, N, K, act);
    else if (cm && !cn)
      gemm_wmma_bf16<true, false><<<grid, 128, 0, stream>>>(Ax, lda, Wx, ldw, bias, Cx, ldc, Cbx, ldcb, M, N, K, act);
    else if (!cm && cn)
      gemm_wmma_bf16<false, true><<<grid, 128, 0, stream>>>(Ax, lda, Wx, ldw, bias, Cx, ldc, Cbx, ldcb, M, N, K, act);
    else
      gemm_wmma_bf16<true, true><<<grid, 128, 0, stream>>>(Ax, lda, Wx, ldw, bias, Cx, ldc, Cbx, ldcb, M, N, K, act);
  };

  // ---- weight bf16 copies ----
  bf16* w_ap_proj = WB(ap_proj_w, 4096);
  bf16* w_fl_proj = WB(fl_proj_w, 8192);
  bf16 *w_k[2][2], *w_q[2][2], *w_v[2][2], *w_out[2][2];
  for (int L = 0; L < 2; L++)
    for (int t = 0; t < 2; t++) {
      w_k[L][t]   = WB(co[L].k_w[t], 16384);
      w_q[L][t]   = WB(co[L].q_w[t], 16384);
      w_v[L][t]   = WB(co[L].v_w[t], 16384);
      w_out[L][t] = WB(co[L].out_w[t], 16384);
    }
  bf16 *w_gih[2], *w_ghh[2];
  for (int l = 0; l < 2; l++) { w_gih[l] = WB(gru_wih[l], 49152); w_ghh[l] = WB(gru_whh[l], 49152); }
  bf16* w_apctx = WB(ap_ctx_w, 16384);
  bf16* w_tgih = WB(tg_wih, 24576);
  bf16* w_tghh = WB(tg_whh, 12288);
  bf16* w_tp   = WB(tp_w, 8192);
  bf16* w_fuse = WB(fuse_w, 49152);
  bf16* w_gate = WB(gate_w, 49152);
  bf16* w_h1 = WB(head1_w, 32768);
  bf16* w_h2 = WB(head2_w, 32768);
  bf16* w_h3 = WB(head3_w, 128);
  bf16* w_c1 = WB(cls1_w, 16384);
  bf16* w_c2 = WB(cls2_w, 768);
  bf16* w_u1 = WB(up1_w, 16384);
  bf16* w_u2 = WB(up2_w, 256);

  // per-relation a_rel/m_rel, transposed per head into W-form bf16
  bf16* relT = ALB((long)2 * 2 * 7 * 4096);  // [layer][a/m][rel][h*1024 + e*32 + d]
  for (int L = 0; L < 2; L++) {
    k_rel_transpose<<<dim3(28), dim3(32, 32), 0, stream>>>(
        Pf + co[L].a_rel[0], relT + (size_t)(L * 2 + 0) * 7 * 4096);
    k_rel_transpose<<<dim3(28), dim3(32, 32), 0, stream>>>(
        Pf + co[L].m_rel[0], relT + (size_t)(L * 2 + 1) * 7 * 4096);
  }

  // ---- activation buffers ----
  bf16* apx_b  = ALB(NAP * 32);
  bf16* flx_b  = ALB(NFL * 64);
  float* tmpA  = ALF(NAP * 128);
  float* tmpF  = ALF(NFL * 128);
  float* x_ap  = ALF(NAP * 128);  bf16* x_ap_b = ALB(NAP * 128);
  float* x_fl0 = ALF(NFL * 128);
  float* x_fl  = ALF(NFL * 128);  bf16* x_fl_b = ALB(NFL * 128);
  float* grot  = ALF(EROT);
  bf16* Kap_b = ALB(NAP * 128);   bf16* Vap_b = ALB(NAP * 128);   float* Qap = ALF(NAP * 128);
  bf16* Kfl_b = ALB(NFL * 128);   bf16* Vfl_b = ALB(NFL * 128);   float* Qfl = ALF(NFL * 128);
  float* KT = ALF(NFL * 128);                                    // shared kt/vt scratch
  const int  relSrcT[7] = {1, 0, 0, 1, 0, 0, 0};  // arr_at, arr_from, cong, dep_from, dep_to, net, rot_aa
  const int  relDstT[7] = {0, 1, 0, 0, 1, 0, 0};
  const long relE[7]    = {100000, 100000, 60000, 100000, 100000, 60000, 60000};
  const int  relIn[7]   = {13, 10, 7, 12, 9, 8, 6};
  float* lgb[7];
  for (int r = 0; r < 7; r++) lgb[r] = ALF(relE[r] * 4);
  float* mx_ap = ALF(NAP * 4);  float* den_ap = ALF(NAP * 4);  float* agg_ap = ALF(NAP * 128);
  float* mx_fl = ALF(NFL * 4);  float* den_fl = ALF(NFL * 4);  float* agg_fl = ALF(NFL * 128);
  bf16* gel_ap = ALB(NAP * 128);  bf16* gel_fl = ALB(NFL * 128);
  float* o_ap = ALF(NAP * 128);   float* o_fl = ALF(NFL * 128);
  bf16* aph_b = ALB(2 * NAP * 128);
  float* gi_ap = ALF(NAP * 384);  float* gh_ap = ALF(NAP * 384);
  bf16* h0_b = ALB(NAP * 128);    bf16* h1_b = ALB(NAP * 128);
  float* apctx_all = ALF(NAP * 128);
  float* ap_context = ALF(NFL * 128);
  float* prevh = ALF(NFL * 64);   bf16* prevh_b = ALB(NFL * 64);
  float* tgi = ALF(NFL * 192);    float* tgh = ALF(NFL * 192);
  float* newh = ALF(NFL * 64);    bf16* newh_b = ALB(NFL * 64);
  float* tailctx = ALF(NFL * 128);
  bf16* cat_b = ALB(NFL * 384);
  float* fusev = ALF(NFL * 128);  float* gatev = ALF(NFL * 128);
  float* flout = ALF(NFL * 128);  bf16* flout_b = ALB(NFL * 128);
  bf16* h1bf = ALB(NFL * 256);    bf16* h2bf = ALB(NFL * 128);
  bf16* c1bf = ALB(NFL * 128);    bf16* u1bf = ALB(NFL * 128);
  float* pred0 = ALF(NFL);        float* upv = ALF(NFL * 2);

  float* out_pred   = (float*)d_out;
  float* out_logits = out_pred + 100000;
  float* out_aph    = out_logits + 600000;
  float* out_tail   = out_aph + 768000;

  // ---- input projections + LN ----
  k_cvt_bf16<<<nb(NAP * 32), 256, 0, stream>>>(ap_x, apx_b, NAP * 32);
  k_cvt_bf16<<<nb(NFL * 64), 256, 0, stream>>>(fl_x, flx_b, NFL * 64);
  GEMM(apx_b, 32, w_ap_proj, 32, Pf + ap_proj_b, tmpA, 128, nullptr, 0, NAP, 128, 32, ACT_NONE);
  k_ln<<<(unsigned)NAP, 32, 0, stream>>>(tmpA, Pf + ap_proj_ln_g, Pf + ap_proj_ln_b, x_ap, x_ap_b);
  GEMM(flx_b, 64, w_fl_proj, 64, Pf + fl_proj_b, tmpF, 128, nullptr, 0, NFL, 128, 64, ACT_NONE);
  k_ln<<<(unsigned)NFL, 32, 0, stream>>>(tmpF, Pf + fl_proj_ln_g, Pf + fl_proj_ln_b, x_fl0, nullptr);

  // ---- gated flight->flight rotation scatter-add (functional, two buffers) ----
  k_rot_gate<<<nb(EROT), 256, 0, stream>>>(rot_ea, Pf + rot1_w, Pf + rot1_b,
                                           Pf + rot2_w, Pf + rot2_b, grot, (int)EROT);
  hipMemcpyAsync(x_fl, x_fl0, (size_t)NFL * 128 * 4, hipMemcpyDeviceToDevice, stream);
  k_rot_scatter<<<nb(EROT * 128), 256, 0, stream>>>(x_fl, x_fl0, grot,
                                                    ei_rot_ff, ei_rot_ff + EROT, (int)EROT);
  k_cvt_bf16<<<nb(NFL * 128), 256, 0, stream>>>(x_fl, x_fl_b, NFL * 128);

  // ---- HGT layers ----
  const float scale = 0.17677669529663687f;  // 1/sqrt(32)
  for (int L = 0; L < 2; L++) {
    GEMM(x_ap_b, 128, w_k[L][0], 128, Pf + co[L].k_b[0], nullptr, 0, Kap_b, 128, NAP, 128, 128, ACT_NONE);
    GEMM(x_ap_b, 128, w_q[L][0], 128, Pf + co[L].q_b[0], Qap, 128, nullptr, 0, NAP, 128, 128, ACT_NONE);
    GEMM(x_ap_b, 128, w_v[L][0], 128, Pf + co[L].v_b[0], nullptr, 0, Vap_b, 128, NAP, 128, 128, ACT_NONE);
    GEMM(x_fl_b, 128, w_k[L][1], 128, Pf + co[L].k_b[1], nullptr, 0, Kfl_b, 128, NFL, 128, 128, ACT_NONE);
    GEMM(x_fl_b, 128, w_q[L][1], 128, Pf + co[L].q_b[1], Qfl, 128, nullptr, 0, NFL, 128, 128, ACT_NONE);
    GEMM(x_fl_b, 128, w_v[L][1], 128, Pf + co[L].v_b[1], nullptr, 0, Vfl_b, 128, NFL, 128, 128, ACT_NONE);

    k_fill<<<nb(NAP * 4), 256, 0, stream>>>(mx_ap, -3.4e38f, NAP * 4);
    k_fill<<<nb(NAP * 4), 256, 0, stream>>>(den_ap, 0.0f, NAP * 4);
    k_fill<<<nb(NAP * 128), 256, 0, stream>>>(agg_ap, 0.0f, NAP * 128);
    k_fill<<<nb(NFL * 4), 256, 0, stream>>>(mx_fl, -3.4e38f, NFL * 4);
    k_fill<<<nb(NFL * 4), 256, 0, stream>>>(den_fl, 0.0f, NFL * 4);
    k_fill<<<nb(NFL * 128), 256, 0, stream>>>(agg_fl, 0.0f, NFL * 128);

    // pass 1: per-relation K transform + logits + segment max
    for (int r = 0; r < 7; r++) {
      long E = relE[r];
      const int* ei = (const int*)d_in[relIn[r]];
      const bf16* Ks = relSrcT[r] ? Kfl_b : Kap_b;
      int Ms = relSrcT[r] ? (int)NFL : (int)NAP;
      for (int h = 0; h < 4; h++)
        GEMM(Ks + 32 * h, 128, relT + ((size_t)(L * 2 + 0) * 7 + r) * 4096 + h * 1024, 32,
             nullptr, KT + 32 * h, 128, nullptr, 0, Ms, 32, 32, ACT_NONE);
      k_edge_logit<<<nb(E * 4), 256, 0, stream>>>(
          ei, ei + E, relDstT[r] ? Qfl : Qap, KT, Pf + co[L].p_rel[r], scale,
          lgb[r], relDstT[r] ? mx_fl : mx_ap, (int)E);
    }
    k_mx_fix<<<nb(NAP * 4), 256, 0, stream>>>(mx_ap, (int)(NAP * 4));
    k_mx_fix<<<nb(NFL * 4), 256, 0, stream>>>(mx_fl, (int)(NFL * 4));

    // pass 2: exp + segment sum
    for (int r = 0; r < 7; r++) {
      long E = relE[r];
      const int* ei = (const int*)d_in[relIn[r]];
      k_edge_exp<<<nb(E * 4), 256, 0, stream>>>(
          ei + E, lgb[r], relDstT[r] ? mx_fl : mx_ap, relDstT[r] ? den_fl : den_ap, (int)E);
    }

    // pass 3: per-relation V transform + weighted aggregation
    for (int r = 0; r < 7; r++) {
      long E = relE[r];
      const int* ei = (const int*)d_in[relIn[r]];
      const bf16* Vs = relSrcT[r] ? Vfl_b : Vap_b;
      int Ms = relSrcT[r] ? (int)NFL : (int)NAP;
      for (int h = 0; h < 4; h++)
        GEMM(Vs + 32 * h, 128, relT + ((size_t)(L * 2 + 1) * 7 + r) * 4096 + h * 1024, 32,
             nullptr, KT + 32 * h, 128, nullptr, 0, Ms, 32, 32, ACT_NONE);
      k_edge_agg<<<nb(E * 128), 256, 0, stream>>>(
          ei, ei + E, lgb[r], relDstT[r] ? den_fl : den_ap, KT,
          relDstT[r] ? agg_fl : agg_ap, (int)E);
    }

    // out linear on gelu(agg), gated skip, residual + LN (in place)
    k_gelu_cvt<<<nb(NAP * 128), 256, 0, stream>>>(agg_ap, gel_ap, NAP * 128);
    GEMM(gel_ap, 128, w_out[L][0], 128, Pf + co[L].out_b[0], o_ap, 128, nullptr, 0, NAP, 128, 128, ACT_NONE);
    k_skip_res_ln<<<(unsigned)NAP, 32, 0, stream>>>(o_ap, x_ap, x_ap_b, Pf + co[L].skip[0],
                                                    Pf + ap_ln_g[L], Pf + ap_ln_b[L]);
    k_gelu_cvt<<<nb(NFL * 128), 256, 0, stream>>>(agg_fl, gel_fl, NFL * 128);
    GEMM(gel_fl, 128, w_out[L][1], 128, Pf + co[L].out_b[1], o_fl, 128, nullptr, 0, NFL, 128, 128, ACT_NONE);
    k_skip_res_ln<<<(unsigned)NFL, 32, 0, stream>>>(o_fl, x_fl, x_fl_b, Pf + co[L].skip[1],
                                                    Pf + fl_ln_g[L], Pf + fl_ln_b[L]);
  }

  // ---- airport GRU (2 layers, single step) -> d_out ap_h_new ----
  k_cvt_bf16<<<nb(2 * NAP * 128), 256, 0, stream>>>(ap_h_in, aph_b, 2 * NAP * 128);
  GEMM(x_ap_b, 128, w_gih[0], 128, Pf + gru_bih[0], gi_ap, 384, nullptr, 0, NAP, 384, 128, ACT_NONE);
  GEMM(aph_b, 128, w_ghh[0], 128, Pf + gru_bhh[0], gh_ap, 384, nullptr, 0, NAP, 384, 128, ACT_NONE);
  k_gru<<<nb(NAP * 128), 256, 0, stream>>>(gi_ap, gh_ap, ap_h_in, out_aph, h0_b, (int)NAP, 128);
  GEMM(h0_b, 128, w_gih[1], 128, Pf + gru_bih[1], gi_ap, 384, nullptr, 0, NAP, 384, 128, ACT_NONE);
  GEMM(aph_b + NAP * 128, 128, w_ghh[1], 128, Pf + gru_bhh[1], gh_ap, 384, nullptr, 0, NAP, 384, 128, ACT_NONE);
  k_gru<<<nb(NAP * 128), 256, 0, stream>>>(gi_ap, gh_ap, ap_h_in + NAP * 128,
                                           out_aph + NAP * 128, h1_b, (int)NAP, 128);
  GEMM(h1_b, 128, w_apctx, 128, Pf + ap_ctx_b, apctx_all, 128, nullptr, 0, NAP, 128, 128, ACT_NONE);
  k_fill<<<nb(NFL * 128), 256, 0, stream>>>(ap_context, 0.0f, NFL * 128);
  k_scatter_rows<<<nb(NFL * 128), 256, 0, stream>>>(ap_context, ei_dep_fr, apctx_all,
                                                    ei_dep_fr + NFL, 128, (int)NFL);

  // ---- tail GRU -> new_tail ----
  k_gather_rows<<<nb(NFL * 64), 256, 0, stream>>>(prevh, tail_table, tail_id, 64, (int)NFL);
  k_cvt_bf16<<<nb(NFL * 64), 256, 0, stream>>>(prevh, prevh_b, NFL * 64);
  GEMM(x_fl_b, 128, w_tgih, 128, Pf + tg_bih, tgi, 192, nullptr, 0, NFL, 192, 128, ACT_NONE);
  GEMM(prevh_b, 64, w_tghh, 64, Pf + tg_bhh, tgh, 192, nullptr, 0, NFL, 192, 64, ACT_NONE);
  k_gru<<<nb(NFL * 64), 256, 0, stream>>>(tgi, tgh, prevh, newh, newh_b, (int)NFL, 64);
  GEMM(newh_b, 64, w_tp, 64, Pf + tp_b, tailctx, 128, nullptr, 0, NFL, 128, 64, ACT_NONE);
  hipMemcpyAsync(out_tail, tail_table, (size_t)NTL * 64 * 4, hipMemcpyDeviceToDevice, stream);
  k_scatter_rows<<<nb(NFL * 64), 256, 0, stream>>>(out_tail, tail_id, newh, nullptr, 64, (int)NFL);

  // ---- fuse gate ----
  k_concat_cvt<<<nb(NFL * 384), 256, 0, stream>>>(x_fl, ap_context, tailctx, cat_b, (int)NFL);
  GEMM(cat_b, 384, w_gate, 384, Pf + gate_b, gatev, 128, nullptr, 0, NFL, 128, 384, ACT_NONE);
  GEMM(cat_b, 384, w_fuse, 384, Pf + fuse_b, fusev, 128, nullptr, 0, NFL, 128, 384, ACT_NONE);
  k_flout<<<(unsigned)NFL, 32, 0, stream>>>(fusev, gatev, x_fl, Pf + fuse_ln_g, Pf + fuse_ln_b,
                                            flout, flout_b);

  // ---- output heads ----
  GEMM(flout_b, 128, w_h1, 128, Pf + head1_b, nullptr, 0, h1bf, 256, NFL, 256, 128, ACT_RELU);
  GEMM(h1bf, 256, w_h2, 256, Pf + head2_b, nullptr, 0, h2bf, 128, NFL, 128, 256, ACT_RELU);
  GEMM(h2bf, 128, w_h3, 128, Pf + head3_b, pred0, 1, nullptr, 0, NFL, 1, 128, ACT_NONE);
  GEMM(flout_b, 128, w_c1, 128, Pf + cls1_b, nullptr, 0, c1bf, 128, NFL, 128, 128, ACT_RELU);
  GEMM(c1bf, 128, w_c2, 128, Pf + cls2_b, out_logits, 6, nullptr, 0, NFL, 6, 128, ACT_NONE);
  GEMM(flout_b, 128, w_u1, 128, Pf + up1_b, nullptr, 0, u1bf, 128, NFL, 128, 128, ACT_RELU);
  GEMM(u1bf, 128, w_u2, 128, Pf + up2_b, upv, 2, nullptr, 0, NFL, 2, 128, ACT_SOFTPLUS);
  k_final<<<nb(NFL), 256, 0, stream>>>(pred0, out_logits, upv, out_pred, (int)NFL);
}
